// CausalSelfAttention_33191507263982
// MI455X (gfx1250) — matrix operand order repaired and checked by reading, not compiled, not run on hardware
//
#include <hip/hip_runtime.h>

// ---------- types for WMMA ----------
typedef __attribute__((ext_vector_type(16))) __bf16         bf16x16;
typedef __attribute__((ext_vector_type(2)))  __bf16         bf16x2;
typedef __attribute__((ext_vector_type(8)))  float          f32x8;
typedef __attribute__((ext_vector_type(2)))  float          f32x2;
typedef __attribute__((ext_vector_type(4)))  unsigned short u16x4;
typedef __attribute__((ext_vector_type(8)))  unsigned short u16x8;
typedef __attribute__((ext_vector_type(16))) unsigned short u16x16;

// native RTNE f32->bf16 (v_cvt_*_bf16_f32 on gfx1250)
__device__ __forceinline__ unsigned short f2bf(float f) {
    return __builtin_bit_cast(unsigned short, (__bf16)f);
}
// packed pair conversion (v_cvt_pk_bf16_f32)
__device__ __forceinline__ unsigned pk2bf(float lo, float hi) {
    f32x2 f = {lo, hi};
    bf16x2 b = __builtin_convertvector(f, bf16x2);
    return __builtin_bit_cast(unsigned, b);
}

__device__ __forceinline__ bf16x16 mkfrag(const unsigned short* p0, const unsigned short* p1) {
    u16x8 a = *(const u16x8*)p0;
    u16x8 b = *(const u16x8*)p1;
    u16x16 r = __builtin_shufflevector(a, b, 0,1,2,3,4,5,6,7,8,9,10,11,12,13,14,15);
    return __builtin_bit_cast(bf16x16, r);
}

__device__ __forceinline__ f32x8 wmma_bf16(bf16x16 a, bf16x16 b, f32x8 c) {
    return __builtin_amdgcn_wmma_f32_16x16x32_bf16(false, a, false, b, (short)0, c, false, false);
}

__device__ __forceinline__ u16x8 pack8(float4 a, float4 b) {
    u16x8 v;
    v[0]=f2bf(a.x); v[1]=f2bf(a.y); v[2]=f2bf(a.z); v[3]=f2bf(a.w);
    v[4]=f2bf(b.x); v[5]=f2bf(b.y); v[6]=f2bf(b.z); v[7]=f2bf(b.w);
    return v;
}
__device__ __forceinline__ u16x4 pack4(float4 a) {
    u16x4 v;
    v[0]=f2bf(a.x); v[1]=f2bf(a.y); v[2]=f2bf(a.z); v[3]=f2bf(a.w);
    return v;
}

// Problem constants
#define BB   2
#define TT   1024
#define CC   2048
#define HH   16
#define DD   128
#define TC   1024
#define KLEN 2048
#define N3C  6144

// =====================================================================
// Kernel 1: qkv = x @ W_qkv  (bf16 WMMA), 128x128 tiles, pipelined staging.
// grid (16, 48), block 256.  Wave w: 2 M-subtiles (rows (w>>1)*32..+31),
// 4 N-subtiles (cols (w&1)*64..+63)  -> 8 WMMAs / K-step.
// =====================================================================
__global__ __launch_bounds__(256) void qkv_gemm_kernel(
        const float* __restrict__ x, const float* __restrict__ w,
        float* __restrict__ qws, float* __restrict__ outk, float* __restrict__ outv) {
    __shared__ __align__(16) unsigned short lds_a[128 * 40];   // [m][k]
    __shared__ __align__(16) unsigned short lds_bt[128 * 40];  // [n][k] transposed

    const int tid  = threadIdx.x;
    const int lane = tid & 31, wave = tid >> 5;
    const int hi = lane >> 4, ln = lane & 15;
    const int m0 = blockIdx.x * 128;
    const int n0 = blockIdx.y * 128;
    const int mbase = (wave >> 1) * 32;
    const int nbase = (wave & 1) * 64;

    // staging thread mapping
    const int ra = tid >> 1, ca = (tid & 1) * 16;          // A: row, 16 k
    const int rb = (tid >> 4) * 2, cb = (tid & 15) * 8;    // B: 2 k-rows, 8 n
    const float* pa  = x + (size_t)(m0 + ra) * CC + ca;
    const float* pb0 = w + (size_t)rb * N3C + n0 + cb;

    float4 A0 = ((const float4*)pa)[0], A1 = ((const float4*)pa)[1];
    float4 A2 = ((const float4*)pa)[2], A3 = ((const float4*)pa)[3];
    float4 B00 = ((const float4*)pb0)[0], B01 = ((const float4*)pb0)[1];
    float4 B10 = ((const float4*)(pb0 + N3C))[0], B11 = ((const float4*)(pb0 + N3C))[1];

    f32x8 acc[2][4] = {};
    for (int k0 = 0; k0 < CC; k0 += 32) {
        {   // store staged tile (bf16) to LDS
            *(u16x8*)&lds_a[ra * 40 + ca]     = pack8(A0, A1);
            *(u16x8*)&lds_a[ra * 40 + ca + 8] = pack8(A2, A3);
            float r0[8] = {B00.x,B00.y,B00.z,B00.w,B01.x,B01.y,B01.z,B01.w};
            float r1[8] = {B10.x,B10.y,B10.z,B10.w,B11.x,B11.y,B11.z,B11.w};
            #pragma unroll
            for (int i = 0; i < 8; ++i)
                *(unsigned*)&lds_bt[(cb + i) * 40 + rb] = pk2bf(r0[i], r1[i]);
        }
        __syncthreads();
        if (k0 + 32 < CC) {     // prefetch next K-tile into registers (overlaps WMMA)
            pa += 32; pb0 += (size_t)32 * N3C;
            A0 = ((const float4*)pa)[0]; A1 = ((const float4*)pa)[1];
            A2 = ((const float4*)pa)[2]; A3 = ((const float4*)pa)[3];
            B00 = ((const float4*)pb0)[0]; B01 = ((const float4*)pb0)[1];
            B10 = ((const float4*)(pb0 + N3C))[0]; B11 = ((const float4*)(pb0 + N3C))[1];
            if (k0 + 64 < CC) {  // L2 prefetch one further tile ahead
                __builtin_prefetch((const void*)(pa + 32), 0, 1);
                __builtin_prefetch((const void*)(pb0 + (size_t)32 * N3C), 0, 1);
            }
        }
        // fragments: batch all DS loads, then 8 WMMAs back-to-back
        bf16x16 afrag[2];
        #pragma unroll
        for (int ms = 0; ms < 2; ++ms) {
            const unsigned short* arow = &lds_a[(mbase + ms * 16 + ln) * 40 + hi * 8];
            afrag[ms] = mkfrag(arow, arow + 16);
        }
        bf16x16 bfrag[4];
        #pragma unroll
        for (int ns = 0; ns < 4; ++ns) {
            const unsigned short* brow = &lds_bt[(nbase + ns * 16 + ln) * 40 + hi * 8];
            bfrag[ns] = mkfrag(brow, brow + 16);
        }
        #pragma unroll
        for (int ms = 0; ms < 2; ++ms)
            #pragma unroll
            for (int ns = 0; ns < 4; ++ns)
                acc[ms][ns] = wmma_bf16(afrag[ms], bfrag[ns], acc[ms][ns]);
        __syncthreads();
    }

    // epilogue: destination class is uniform per block
    const int s = n0 >> 11;   // 0=q, 1=k, 2=v
    const int nloc = (n0 & 2047) + nbase;
    if (s == 0) {
        const float qscale = 0.08838834764831845f;  // 1/sqrt(128)
        #pragma unroll
        for (int ms = 0; ms < 2; ++ms)
        #pragma unroll
        for (int ns = 0; ns < 4; ++ns) {
            int n = nloc + ns * 16 + ln, h = n >> 7, d = n & 127;
            #pragma unroll
            for (int r = 0; r < 8; ++r) {
                int m = m0 + mbase + ms * 16 + r + hi * 8;
                int b = m >> 10, t = m & 1023;
                qws[((size_t)(b*HH + h) * TT + t) * DD + d] = acc[ms][ns][r] * qscale;
            }
        }
    } else {
        float* o = (s == 1) ? outk : outv;
        #pragma unroll
        for (int ms = 0; ms < 2; ++ms)
        #pragma unroll
        for (int ns = 0; ns < 4; ++ns) {
            int n = nloc + ns * 16 + ln, h = n >> 7, d = n & 127;
            #pragma unroll
            for (int r = 0; r < 8; ++r) {
                int m = m0 + mbase + ms * 16 + r + hi * 8;
                int b = m >> 10, t = m & 1023;
                o[((size_t)(b*HH + h) * KLEN + TC + t) * DD + d] = acc[ms][ns][r];
            }
        }
    }
}

// =====================================================================
// Kernel 2: copy cached k/v into rows [0,TC) of the output caches
// grid 4096, block 256
// =====================================================================
__global__ __launch_bounds__(256) void cache_copy_kernel(
        const float* __restrict__ ck, const float* __restrict__ cv,
        float* __restrict__ ok, float* __restrict__ ov) {
    size_t e = ((size_t)blockIdx.x * 256 + threadIdx.x) * 4;
    size_t bh = e >> 17;
    size_t rem = e & 131071;
    float4 k4 = *(const float4*)(ck + e);
    float4 v4 = *(const float4*)(cv + e);
    *(float4*)(ok + bh * (size_t)(KLEN * DD) + rem) = k4;
    *(float4*)(ov + bh * (size_t)(KLEN * DD) + rem) = v4;
}

// =====================================================================
// Kernel 3: flash attention.  grid (B*H, T/128), block 256 (8 waves x 16 q rows)
// =====================================================================
__global__ __launch_bounds__(256) void attn_kernel(
        const float* __restrict__ qws, const float* __restrict__ kc,
        const float* __restrict__ vc, float* __restrict__ aws,
        const int* __restrict__ seq_start_p) {
    __shared__ __align__(16) unsigned short lds_k [64  * 136]; // [key][d]
    __shared__ __align__(16) unsigned short lds_vt[128 * 72];  // [d][key]
    __shared__ __align__(16) unsigned short lds_p [8 * 16 * 72]; // per-wave P [m][key]

    const int tid  = threadIdx.x;
    const int lane = tid & 31, wave = tid >> 5;
    const int hi = lane >> 4, ln = lane & 15;
    const int bh = blockIdx.x;
    const int b = bh >> 4, h = bh & 15;
    const int qt = blockIdx.y;
    const int seq0 = seq_start_p[0];

    // resident q fragments (16 rows x 128 d), scale pre-folded
    const float* qrow = qws + ((size_t)bh * TT + qt * 128 + wave * 16 + ln) * DD;
    bf16x16 qf[4];
    #pragma unroll
    for (int kcn = 0; kcn < 4; ++kcn) {
        const int c0 = kcn * 32 + hi * 8;
        float t0[8], t1[8];
        *(float4*)&t0[0] = *(const float4*)(qrow + c0);
        *(float4*)&t0[4] = *(const float4*)(qrow + c0 + 4);
        *(float4*)&t1[0] = *(const float4*)(qrow + c0 + 16);
        *(float4*)&t1[4] = *(const float4*)(qrow + c0 + 20);
        u16x16 u = {};
        #pragma unroll
        for (int i = 0; i < 8; ++i) { u[i] = f2bf(t0[i]); u[i + 8] = f2bf(t1[i]); }
        qf[kcn] = __builtin_bit_cast(bf16x16, u);
    }

    f32x8 oacc[8] = {};
    float mrun[8], lrun[8];
    #pragma unroll
    for (int r = 0; r < 8; ++r) { mrun[r] = -3.0e38f; lrun[r] = 0.0f; }

    const float* kbase = kc + (size_t)bh * KLEN * DD;
    const float* vbase = vc + (size_t)bh * KLEN * DD;
    int jmax = (seq0 + qt * 128 + 127) >> 6;
    if (jmax > (KLEN / 64) - 1) jmax = (KLEN / 64) - 1;

    // staging maps
    const int kk = tid >> 5, kd = (tid & 31) * 4;          // K: key=kk(+8*it), 4 d
    const int vk = (tid >> 5) * 2, vd = (tid & 31) * 4;    // V: 2 keys, 4 d

    for (int j = 0; j <= jmax; ++j) {
        __syncthreads();
        #pragma unroll
        for (int it = 0; it < 8; ++it) {   // K tile: 64 keys x 128 d -> [key][d]
            int key = kk + it * 8;
            float4 k4 = *(const float4*)(kbase + (size_t)(j * 64 + key) * DD + kd);
            *(u16x4*)&lds_k[key * 136 + kd] = pack4(k4);
        }
        #pragma unroll
        for (int it = 0; it < 4; ++it) {   // V tile transposed: [d][key], packed key-pairs
            int key = vk + it * 16;
            const float* v0 = vbase + (size_t)(j * 64 + key) * DD + vd;
            float4 a = *(const float4*)v0;
            float4 c = *(const float4*)(v0 + DD);
            *(unsigned*)&lds_vt[(vd + 0) * 72 + key] = pk2bf(a.x, c.x);
            *(unsigned*)&lds_vt[(vd + 1) * 72 + key] = pk2bf(a.y, c.y);
            *(unsigned*)&lds_vt[(vd + 2) * 72 + key] = pk2bf(a.z, c.z);
            *(unsigned*)&lds_vt[(vd + 3) * 72 + key] = pk2bf(a.w, c.w);
        }
        __syncthreads();
        if (j < jmax) {  // L2 prefetch of next key tile
            __builtin_prefetch((const void*)(kbase + (size_t)((j + 1) * 64 + kk) * DD + kd), 0, 1);
            __builtin_prefetch((const void*)(vbase + (size_t)((j + 1) * 64 + kk) * DD + kd), 0, 1);
        }

        // S = Q K^T : per 16-key sub-block, batch 4 fragment loads then 4 WMMAs
        f32x8 s[4];
        #pragma unroll
        for (int jj = 0; jj < 4; ++jj) {
            bf16x16 kf[4];
            #pragma unroll
            for (int kcn = 0; kcn < 4; ++kcn) {
                const unsigned short* kr = &lds_k[(jj * 16 + ln) * 136 + kcn * 32 + hi * 8];
                kf[kcn] = mkfrag(kr, kr + 16);
            }
            f32x8 c = {};
            #pragma unroll
            for (int kcn = 0; kcn < 4; ++kcn) c = wmma_bf16(qf[kcn], kf[kcn], c);
            s[jj] = c;
        }

        // online softmax per row; write P (bf16) into per-wave LDS
        unsigned short* pwave = &lds_p[wave * 16 * 72];
        #pragma unroll
        for (int r = 0; r < 8; ++r) {
            const int trow = qt * 128 + wave * 16 + r + hi * 8;
            const int qabs = seq0 + trow;
            float sv[4], rowm = -3.0e38f;
            #pragma unroll
            for (int jj = 0; jj < 4; ++jj) {
                int kpos = j * 64 + jj * 16 + ln;
                float v = s[jj][r];
                v = (kpos <= qabs) ? v : -3.0e38f;
                sv[jj] = v;
                rowm = fmaxf(rowm, v);
            }
            #pragma unroll
            for (int mm = 8; mm >= 1; mm >>= 1) rowm = fmaxf(rowm, __shfl_xor(rowm, mm, 32));
            float newm = fmaxf(mrun[r], rowm);
            float corr = __expf(mrun[r] - newm);
            float psum = 0.0f;
            unsigned short* pr = pwave + (r + hi * 8) * 72;
            #pragma unroll
            for (int jj = 0; jj < 4; ++jj) {
                float p = __expf(sv[jj] - newm);
                psum += p;
                pr[jj * 16 + ln] = f2bf(p);
            }
            #pragma unroll
            for (int mm = 8; mm >= 1; mm >>= 1) psum += __shfl_xor(psum, mm, 32);
            lrun[r] = lrun[r] * corr + psum;
            mrun[r] = newm;
            #pragma unroll
            for (int ds = 0; ds < 8; ++ds) oacc[ds][r] *= corr;
        }

        // O += P @ V : per K-chunk, batch 4 V-fragment loads, then 4 WMMAs
        #pragma unroll
        for (int kcn = 0; kcn < 2; ++kcn) {
            const unsigned short* ar = pwave + ln * 72 + kcn * 32 + hi * 8;
            bf16x16 af = mkfrag(ar, ar + 16);
            #pragma unroll
            for (int dsg = 0; dsg < 2; ++dsg) {
                bf16x16 vf[4];
                #pragma unroll
                for (int i = 0; i < 4; ++i) {
                    const unsigned short* vr =
                        &lds_vt[((dsg * 4 + i) * 16 + ln) * 72 + kcn * 32 + hi * 8];
                    vf[i] = mkfrag(vr, vr + 16);
                }
                #pragma unroll
                for (int i = 0; i < 4; ++i)
                    oacc[dsg * 4 + i] = wmma_bf16(af, vf[i], oacc[dsg * 4 + i]);
            }
        }
    }

    // normalize and store to (B,T,C) workspace for the projection
    #pragma unroll
    for (int r = 0; r < 8; ++r) {
        float inv = 1.0f / lrun[r];
        int trow = qt * 128 + wave * 16 + r + hi * 8;
        float* orow = aws + ((size_t)b * TT + trow) * CC + h * DD;
        #pragma unroll
        for (int ds = 0; ds < 8; ++ds) orow[ds * 16 + ln] = oacc[ds][r] * inv;
    }
}

// =====================================================================
// Kernel 4: out = attn @ W_proj + b_proj.  128x128 tiles, grid (16, 16)
// =====================================================================
__global__ __launch_bounds__(256) void proj_gemm_kernel(
        const float* __restrict__ a, const float* __restrict__ w,
        const float* __restrict__ bias, float* __restrict__ out) {
    __shared__ __align__(16) unsigned short lds_a[128 * 40];
    __shared__ __align__(16) unsigned short lds_bt[128 * 40];

    const int tid  = threadIdx.x;
    const int lane = tid & 31, wave = tid >> 5;
    const int hi = lane >> 4, ln = lane & 15;
    const int m0 = blockIdx.x * 128;
    const int n0 = blockIdx.y * 128;
    const int mbase = (wave >> 1) * 32;
    const int nbase = (wave & 1) * 64;

    const int ra = tid >> 1, ca = (tid & 1) * 16;
    const int rb = (tid >> 4) * 2, cb = (tid & 15) * 8;
    const float* pa  = a + (size_t)(m0 + ra) * CC + ca;
    const float* pb0 = w + (size_t)rb * CC + n0 + cb;

    float4 A0 = ((const float4*)pa)[0], A1 = ((const float4*)pa)[1];
    float4 A2 = ((const float4*)pa)[2], A3 = ((const float4*)pa)[3];
    float4 B00 = ((const float4*)pb0)[0], B01 = ((const float4*)pb0)[1];
    float4 B10 = ((const float4*)(pb0 + CC))[0], B11 = ((const float4*)(pb0 + CC))[1];

    f32x8 acc[2][4] = {};
    for (int k0 = 0; k0 < CC; k0 += 32) {
        {
            *(u16x8*)&lds_a[ra * 40 + ca]     = pack8(A0, A1);
            *(u16x8*)&lds_a[ra * 40 + ca + 8] = pack8(A2, A3);
            float r0[8] = {B00.x,B00.y,B00.z,B00.w,B01.x,B01.y,B01.z,B01.w};
            float r1[8] = {B10.x,B10.y,B10.z,B10.w,B11.x,B11.y,B11.z,B11.w};
            #pragma unroll
            for (int i = 0; i < 8; ++i)
                *(unsigned*)&lds_bt[(cb + i) * 40 + rb] = pk2bf(r0[i], r1[i]);
        }
        __syncthreads();
        if (k0 + 32 < CC) {
            pa += 32; pb0 += (size_t)32 * CC;
            A0 = ((const float4*)pa)[0]; A1 = ((const float4*)pa)[1];
            A2 = ((const float4*)pa)[2]; A3 = ((const float4*)pa)[3];
            B00 = ((const float4*)pb0)[0]; B01 = ((const float4*)pb0)[1];
            B10 = ((const float4*)(pb0 + CC))[0]; B11 = ((const float4*)(pb0 + CC))[1];
            if (k0 + 64 < CC) {
                __builtin_prefetch((const void*)(pa + 32), 0, 1);
                __builtin_prefetch((const void*)(pb0 + (size_t)32 * CC), 0, 1);
            }
        }
        bf16x16 afrag[2];
        #pragma unroll
        for (int ms = 0; ms < 2; ++ms) {
            const unsigned short* arow = &lds_a[(mbase + ms * 16 + ln) * 40 + hi * 8];
            afrag[ms] = mkfrag(arow, arow + 16);
        }
        bf16x16 bfrag[4];
        #pragma unroll
        for (int ns = 0; ns < 4; ++ns) {
            const unsigned short* brow = &lds_bt[(nbase + ns * 16 + ln) * 40 + hi * 8];
            bfrag[ns] = mkfrag(brow, brow + 16);
        }
        #pragma unroll
        for (int ms = 0; ms < 2; ++ms)
            #pragma unroll
            for (int ns = 0; ns < 4; ++ns)
                acc[ms][ns] = wmma_bf16(afrag[ms], bfrag[ns], acc[ms][ns]);
        __syncthreads();
    }
    #pragma unroll
    for (int ms = 0; ms < 2; ++ms)
    #pragma unroll
    for (int ns = 0; ns < 4; ++ns) {
        int n = n0 + nbase + ns * 16 + ln;
        float bz = bias[n];
        #pragma unroll
        for (int r = 0; r < 8; ++r) {
            int m = m0 + mbase + ms * 16 + r + hi * 8;
            out[(size_t)m * CC + n] = acc[ms][ns][r] + bz;
        }
    }
}

// =====================================================================
extern "C" void kernel_launch(void* const* d_in, const int* in_sizes, int n_in,
                              void* d_out, int out_size, void* d_ws, size_t ws_size,
                              hipStream_t stream) {
    const float* x       = (const float*)d_in[0];
    const float* cache_k = (const float*)d_in[1];
    const float* cache_v = (const float*)d_in[2];
    const float* W_qkv   = (const float*)d_in[3];
    const float* W_proj  = (const float*)d_in[4];
    const float* b_proj  = (const float*)d_in[5];
    const int*   seq_st  = (const int*)d_in[6];

    float* out   = (float*)d_out;
    float* out_k = out + (size_t)BB * TT * CC;
    float* out_v = out_k + (size_t)BB * HH * KLEN * DD;

    float* q_ws = (float*)d_ws;                              // (B,H,T,D) f32
    float* a_ws = q_ws + (size_t)BB * HH * TT * DD;          // (B,T,C)   f32

    hipLaunchKernelGGL(qkv_gemm_kernel, dim3(16, 48), dim3(256), 0, stream,
                       x, W_qkv, q_ws, out_k, out_v);
    hipLaunchKernelGGL(cache_copy_kernel, dim3(4096), dim3(256), 0, stream,
                       cache_k, cache_v, out_k, out_v);
    hipLaunchKernelGGL(attn_kernel, dim3(BB * HH, TT / 128), dim3(256), 0, stream,
                       q_ws, out_k, out_v, a_ws, seq_st);
    hipLaunchKernelGGL(proj_gemm_kernel, dim3(16, 16), dim3(256), 0, stream,
                       a_ws, W_proj, b_proj, out);
}
